// Model_57475252355559
// MI455X (gfx1250) — matrix operand order repaired and checked by reading, not compiled, not run on hardware
//
#include <hip/hip_runtime.h>
#include <cstdint>

// ============================================================================
// PointMLP-style network for MI455X (gfx1250, wave32).
//
// All conv1x1 / linear layers -> one fused WMMA GEMM kernel
//   (v_wmma_f32_16x16x32_f16, f16 in / f32 accumulate, fused bias+BN+res+ReLU).
//
// Data layout: activations are point-major f16  act[point][channel], channel
// count always padded to a multiple of 32.  Weights are pre-converted once to
// f16 [Mpad(64)][Kpad(32)] zero-padded.  Consequences:
//   * tile staging global->LDS is pure b128 traffic (async-to-LDS via the
//     gfx1250 GLOBAL_LOAD_ASYNC_TO_LDS_B128 path, ds stores otherwise)
//   * WMMA A operand = 2x ds_load_b128 (this code uses contiguous K = h*16 + i
//     in both A and B), B operand = 2x ds_load_b128
//   * epilogue = 1x global_store_b128 per lane (8 consecutive channels)
// ============================================================================

typedef __attribute__((ext_vector_type(16))) _Float16 v16h;
typedef __attribute__((ext_vector_type(8)))  _Float16 v8h;
typedef __attribute__((ext_vector_type(8)))  float    v8f;
typedef int v4i __attribute__((vector_size(16)));
typedef _Float16 h16;

#define BATCH 16
#define NPTS  2048
#define KNN_K 32

#if defined(__has_builtin)
#if __has_builtin(__builtin_amdgcn_global_load_async_to_lds_b128) && \
    __has_builtin(__builtin_amdgcn_s_wait_asynccnt)
#define ASYNC_COPY 1
#endif
#endif
#ifndef ASYNC_COPY
#define ASYNC_COPY 0
#endif

#if ASYNC_COPY
typedef __attribute__((address_space(1))) v4i glb_v4i;
typedef __attribute__((address_space(3))) v4i lds_v4i;
#endif

// ---------------------------------------------------------------------------
// Weight conversion: f32 [M][K] -> f16 [Mpad][Kpad], zero-padded.
// ---------------------------------------------------------------------------
__global__ void k_cvt_pad(const float* __restrict__ W, h16* __restrict__ O,
                          int M, int K, int Kp, long long tot)
{
  long long t = (long long)blockIdx.x * blockDim.x + threadIdx.x;
  if (t >= tot) return;
  int m = (int)(t / Kp), k = (int)(t % Kp);
  O[t] = (m < M && k < K) ? (h16)W[(size_t)m*K + k] : (h16)0.f;
}

// ---------------------------------------------------------------------------
// Embedding: two 3->16 convs + BN + ReLU -> data[point][32] (f16)
// ---------------------------------------------------------------------------
__global__ void k_embed(const float* __restrict__ pts, const float* __restrict__ nrm,
                        const float* __restrict__ W1, const float* __restrict__ g1,
                        const float* __restrict__ b1, const float* __restrict__ m1,
                        const float* __restrict__ v1,
                        const float* __restrict__ W2, const float* __restrict__ g2,
                        const float* __restrict__ b2, const float* __restrict__ m2,
                        const float* __restrict__ v2,
                        h16* __restrict__ out, int P)
{
  int p = blockIdx.x * blockDim.x + threadIdx.x;
  if (p >= P) return;
  float x0 = pts[p*3+0], x1 = pts[p*3+1], x2 = pts[p*3+2];
  float n0 = nrm[p*3+0], n1 = nrm[p*3+1], n2 = nrm[p*3+2];
  #pragma unroll
  for (int c = 0; c < 16; ++c) {
    float a = W1[c*3+0]*x0 + W1[c*3+1]*x1 + W1[c*3+2]*x2;
    float s = g1[c] * rsqrtf(v1[c] + 1e-5f);
    a = fmaxf(a*s + (b1[c] - m1[c]*s), 0.f);
    out[(size_t)p*32 + c] = (h16)a;
    float a2 = W2[c*3+0]*n0 + W2[c*3+1]*n1 + W2[c*3+2]*n2;
    float s2 = g2[c] * rsqrtf(v2[c] + 1e-5f);
    a2 = fmaxf(a2*s2 + (b2[c] - m2[c]*s2), 0.f);
    out[(size_t)p*32 + 16 + c] = (h16)a2;
  }
}

// ---------------------------------------------------------------------------
// Farthest point sampling: one block per batch.
// ---------------------------------------------------------------------------
__global__ void k_fps(const float* __restrict__ xyz, int Ncur, int S,
                      int* __restrict__ fpsIdx)
{
  int b = blockIdx.x;
  const float* base = xyz + (size_t)b * Ncur * 3;
  __shared__ float dist[NPTS];
  __shared__ float rs[256];
  __shared__ int   ri[256];
  __shared__ int   sfar;
  for (int n = threadIdx.x; n < Ncur; n += 256) dist[n] = 1e10f;
  if (threadIdx.x == 0) sfar = 0;
  __syncthreads();
  for (int i = 0; i < S; ++i) {
    int far = sfar;
    if (threadIdx.x == 0) fpsIdx[b*S + i] = far;
    float fx = base[far*3+0], fy = base[far*3+1], fz = base[far*3+2];
    float bm = -1.f; int ba = 0;
    for (int n = threadIdx.x; n < Ncur; n += 256) {
      float dx = base[n*3+0]-fx, dy = base[n*3+1]-fy, dz = base[n*3+2]-fz;
      float d  = dx*dx + dy*dy + dz*dz;
      float dn = fminf(dist[n], d);
      dist[n] = dn;
      if (dn > bm) { bm = dn; ba = n; }
    }
    rs[threadIdx.x] = bm; ri[threadIdx.x] = ba;
    __syncthreads();
    for (int o = 128; o > 0; o >>= 1) {
      if (threadIdx.x < o && rs[threadIdx.x+o] > rs[threadIdx.x]) {
        rs[threadIdx.x] = rs[threadIdx.x+o]; ri[threadIdx.x] = ri[threadIdx.x+o];
      }
      __syncthreads();
    }
    if (threadIdx.x == 0) sfar = ri[0];
    __syncthreads();
  }
}

__global__ void k_gather_xyz(const float* __restrict__ xyz, const int* __restrict__ fpsIdx,
                             float* __restrict__ nxyz, int S, int Ncur)
{
  int t = blockIdx.x * blockDim.x + threadIdx.x;
  if (t >= BATCH * S) return;
  int b = t / S;
  int id = fpsIdx[t];
  const float* p = xyz + ((size_t)b*Ncur + id) * 3;
  nxyz[t*3+0] = p[0]; nxyz[t*3+1] = p[1]; nxyz[t*3+2] = p[2];
}

// ---------------------------------------------------------------------------
// KNN: thread per query, 32 smallest squared distances (order-invariant use).
// ---------------------------------------------------------------------------
__global__ void k_knn(const float* __restrict__ xyz, const float* __restrict__ nxyz,
                      int* __restrict__ knn, int S, int Ncur)
{
  int t = blockIdx.x * blockDim.x + threadIdx.x;
  if (t >= BATCH * S) return;
  int b = t / S;
  float qx = nxyz[t*3+0], qy = nxyz[t*3+1], qz = nxyz[t*3+2];
  float kd[KNN_K]; int ki[KNN_K];
  for (int j = 0; j < KNN_K; ++j) { kd[j] = 3e38f; ki[j] = 0; }
  const float* base = xyz + (size_t)b * Ncur * 3;
  for (int n = 0; n < Ncur; ++n) {
    float dx = base[n*3+0]-qx, dy = base[n*3+1]-qy, dz = base[n*3+2]-qz;
    float d  = dx*dx + dy*dy + dz*dz;
    if (d < kd[KNN_K-1]) {
      int j = KNN_K-1;
      while (j > 0 && kd[j-1] > d) { kd[j] = kd[j-1]; ki[j] = ki[j-1]; --j; }
      kd[j] = d; ki[j] = n;
    }
  }
  for (int j = 0; j < KNN_K; ++j) knn[(size_t)t*KNN_K + j] = ki[j];
}

// ---------------------------------------------------------------------------
// Group gather: graw[g][c][kk] (c in [0,C+3)), plus per-(g,c) mean over k.
// feat is point-major with stride C.
// ---------------------------------------------------------------------------
__global__ void k_group_gather(const h16* __restrict__ feat, const float* __restrict__ xyz,
                               const int* __restrict__ knn,
                               float* __restrict__ graw, float* __restrict__ gmean,
                               int S, int C, int Ncur)
{
  int Cp = C + 3;
  long long tot = (long long)BATCH * S * Cp;
  long long t = (long long)blockIdx.x * blockDim.x + threadIdx.x;
  if (t >= tot) return;
  int g = (int)(t / Cp), c = (int)(t % Cp);
  int b = g / S;
  const int* id = knn + (long long)g * KNN_K;
  long long ob = t * KNN_K;
  float sum = 0.f;
  for (int kk = 0; kk < KNN_K; ++kk) {
    int n = id[kk];
    float v = (c < C) ? (float)feat[((long long)b*Ncur + n)*C + c]
                      : xyz[((long long)b*Ncur + n)*3 + (c - C)];
    graw[ob + kk] = v;
    sum += v;
  }
  gmean[t] = sum * (1.f / KNN_K);
}

__global__ void k_fill(float* p, int n)
{
  int t = blockIdx.x * blockDim.x + threadIdx.x;
  if (t < n) p[t] = 0.f;
}

// Per-batch sum / sumsq of (graw - group-mean).  Mb % 256 == 0 always.
__global__ void k_stats(const float* __restrict__ graw, const float* __restrict__ gmean,
                        float* __restrict__ stats, long long Mb)
{
  long long t = (long long)blockIdx.x * blockDim.x + threadIdx.x;
  int b = (int)(t / Mb);
  float v = graw[t] - gmean[t >> 5];
  __shared__ float ss[256], sq[256];
  ss[threadIdx.x] = v; sq[threadIdx.x] = v * v;
  __syncthreads();
  for (int o = 128; o > 0; o >>= 1) {
    if (threadIdx.x < o) { ss[threadIdx.x] += ss[threadIdx.x+o]; sq[threadIdx.x] += sq[threadIdx.x+o]; }
    __syncthreads();
  }
  if (threadIdx.x == 0) { atomicAdd(&stats[b*2+0], ss[0]); atomicAdd(&stats[b*2+1], sq[0]); }
}

// Normalize + anchor concat -> Xg point-major [G*32][dchPad], zero K-padding.
__global__ void k_norm_concat(const float* __restrict__ graw, const float* __restrict__ gmean,
                              const float* __restrict__ stats,
                              const float* __restrict__ alpha, const float* __restrict__ beta,
                              const h16* __restrict__ feat, const int* __restrict__ fpsIdx,
                              h16* __restrict__ Xg, int S, int C, int Ncur,
                              long long Mb, int dchPad)
{
  int Cp = C + 3, dch = 2*C + 3, G = BATCH * S;
  long long tot = (long long)G * KNN_K * dchPad;
  long long t = (long long)blockIdx.x * blockDim.x + threadIdx.x;
  if (t >= tot) return;
  int c = (int)(t % dchPad);
  long long p = t / dchPad;                 // point = g*32 + kk
  int g = (int)(p >> 5), kk = (int)(p & 31);
  int b = g / S, s = g % S;
  float v = 0.f;
  if (c < Cp) {
    long long gi = (long long)g * Cp + c;
    float mean = gmean[gi];
    float sum = stats[b*2+0], sq = stats[b*2+1];
    float var = (sq - sum*sum/(float)Mb) / (float)(Mb - 1);
    float stdv = sqrtf(fmaxf(var, 0.f));
    v = alpha[c] * ((graw[gi*KNN_K + kk] - mean) / (stdv + 1e-5f)) + beta[c];
  } else if (c < dch) {
    int cc = c - Cp;                        // anchor channel
    v = (float)feat[((long long)b*Ncur + fpsIdx[b*S + s])*C + cc];
  }
  Xg[t] = (h16)v;
}

// ---------------------------------------------------------------------------
// Fused WMMA GEMM:  Y[n][m] = act( BN( sum_k W[m][k]*X[n][k] + bias ) + res )
// 8 wave32 / block: 4 M-tiles x 2 N-tiles = 64x32 per block, K step 32,
// double-buffered LDS, async global->LDS (GLOBAL_LOAD_ASYNC_TO_LDS_B128).
// ---------------------------------------------------------------------------
#define LPAD 40   // LDS row pitch in halfs (80B: 16 distinct banks across lanes)
__global__ void k_gemm(const h16* __restrict__ Wh, const h16* __restrict__ X,
                       h16* __restrict__ Y,
                       const float* __restrict__ bias,
                       const float* __restrict__ bng, const float* __restrict__ bnb,
                       const float* __restrict__ bnm, const float* __restrict__ bnv,
                       const h16* __restrict__ res,
                       int Cout, int Kst, int N, int relu)
{
  __shared__ h16 Ws[2][64*LPAD];
  __shared__ h16 Xs[2][32*LPAD];
  int tid  = threadIdx.x, lane = tid & 31, wave = tid >> 5;
  int mw = wave >> 1, nw = wave & 1;
  int h  = lane >> 4, low = lane & 15;
  int nblk = blockIdx.x * 32;
  int mblk = blockIdx.y * 64;

  int wrow = tid >> 2, wseg = tid & 3;              // W: 64 rows x 4 b128 segs
  int xrow = (tid & 127) >> 2, xseg = tid & 3;      // X: threads 0..127
  bool doX = tid < 128;

  auto stage = [&](int buf, int k0) {
    const h16* wsrc = Wh + (size_t)(mblk + wrow)*Kst + k0 + wseg*8;
    h16*       wdst = &Ws[buf][wrow*LPAD + wseg*8];
    const h16* xsrc = X + (size_t)(nblk + xrow)*Kst + k0 + xseg*8;
    h16*       xdst = &Xs[buf][xrow*LPAD + xseg*8];
#if ASYNC_COPY
    __builtin_amdgcn_global_load_async_to_lds_b128(
        (glb_v4i*)(uintptr_t)wsrc, (lds_v4i*)(uint32_t)(uintptr_t)wdst, 0, 0);
    if (doX)
      __builtin_amdgcn_global_load_async_to_lds_b128(
          (glb_v4i*)(uintptr_t)xsrc, (lds_v4i*)(uint32_t)(uintptr_t)xdst, 0, 0);
#else
    *(v4i*)wdst = *(const v4i*)wsrc;
    if (doX) *(v4i*)xdst = *(const v4i*)xsrc;
    __builtin_prefetch(wsrc + 32, 0, 1);            // next K slab
#endif
  };

  stage(0, 0);
  int nk = Kst >> 5;
  v8f acc = {};
  for (int ks = 0; ks < nk; ++ks) {
    int buf = ks & 1;
#if ASYNC_COPY
    __builtin_amdgcn_s_wait_asynccnt(0);
#endif
    __syncthreads();
    if (ks + 1 < nk) stage(buf ^ 1, (ks + 1) << 5);

    // A operand: W row (M = mw*16+low); K = h*16 + i, matching B
    // -> 16 contiguous halfs -> 2x ds_load_b128.
    const h16* arow = &Ws[buf][(mw*16 + low)*LPAD];
    v8h a0 = *(const v8h*)(arow + h*16);
    v8h a1 = *(const v8h*)(arow + h*16 + 8);
    // B operand: X point (N = nw*16+low), K = h*16 + i -> 16 contiguous halfs.
    const h16* brow = &Xs[buf][(nw*16 + low)*LPAD + h*16];
    v8h b0 = *(const v8h*)(brow + 0);
    v8h b1 = *(const v8h*)(brow + 8);
    v16h amat = __builtin_shufflevector(a0, a1, 0,1,2,3,4,5,6,7,8,9,10,11,12,13,14,15);
    v16h bmat = __builtin_shufflevector(b0, b1, 0,1,2,3,4,5,6,7,8,9,10,11,12,13,14,15);
    acc = __builtin_amdgcn_wmma_f32_16x16x32_f16(false, amat, false, bmat,
                                                 (short)0, acc, false, false);
  }

  // Epilogue: lane owns 8 consecutive channels of one point -> b128 store.
  int n  = nblk + nw*16 + low;
  int m0 = mblk + mw*16 + (h << 3);
  if (m0 + 8 <= Cout) {
    v8h rv = {};
    if (res) rv = *(const v8h*)&res[(size_t)n*Cout + m0];
    v8h pack;
    #pragma unroll
    for (int r = 0; r < 8; ++r) {
      int m = m0 + r;
      float y = acc[r];
      if (bias) y += bias[m];
      if (bng) { float sc = bng[m]*rsqrtf(bnv[m]+1e-5f); y = y*sc + (bnb[m]-bnm[m]*sc); }
      if (res) y += (float)rv[r];
      if (relu) y = fmaxf(y, 0.f);
      pack[r] = (h16)y;
    }
    *(v8h*)&Y[(size_t)n*Cout + m0] = pack;
  } else {
    for (int r = 0; r < 8; ++r) {
      int m = m0 + r;
      if (m < Cout) {
        float y = acc[r];
        if (bias) y += bias[m];
        if (bng) { float sc = bng[m]*rsqrtf(bnv[m]+1e-5f); y = y*sc + (bnb[m]-bnm[m]*sc); }
        if (res) y += (float)res[(size_t)n*Cout + m];
        if (relu) y = fmaxf(y, 0.f);
        Y[(size_t)n*Cout + m] = (h16)y;
      }
    }
  }
}

// Max over k=32 within each group: [(g*32+kk)][Cout] -> [g][Cout]
__global__ void k_maxpool(const h16* __restrict__ X, h16* __restrict__ Y,
                          int Cout, long long G)
{
  long long t = (long long)blockIdx.x * blockDim.x + threadIdx.x;
  if (t >= (long long)Cout * G) return;
  long long g = t / Cout; int c = (int)(t % Cout);
  const h16* p = X + (g*KNN_K)*(long long)Cout + c;
  float m = -3e38f;
  for (int kk = 0; kk < KNN_K; ++kk) m = fmaxf(m, (float)p[(long long)kk*Cout]);
  Y[g*Cout + c] = (h16)m;
}

// Final mean over points per batch -> out f32 [B][3]; H is [point][3].
__global__ void k_mean(const h16* __restrict__ H, float* __restrict__ out, int Pn)
{
  int t = blockIdx.x * blockDim.x + threadIdx.x;
  if (t >= BATCH * 3) return;
  int b = t / 3, o = t % 3;
  float s = 0.f;
  for (int j = 0; j < Pn; ++j)
    s += (float)H[((size_t)b*Pn + j)*3 + o];
  out[b*3 + o] = s / (float)Pn;
}

// ===========================================================================
// Driver
// ===========================================================================
struct BNp  { const float *b, *g, *m, *v; };
struct CBRp { BNp bn; const float* W; };
struct Resp { CBRp n1, n2; };

extern "C" void kernel_launch(void* const* d_in, const int* in_sizes, int n_in,
                              void* d_out, int out_size, void* d_ws, size_t ws_size,
                              hipStream_t stream)
{
  (void)in_sizes; (void)n_in; (void)out_size; (void)ws_size;

  // ---- input walk: jax pytree leaf order (dict keys sorted) ---------------
  int ip = 0;
  auto F = [&]() { return (const float*)d_in[ip++]; };
  const float* normals = F();
  BNp cbn1{F(), F(), F(), F()};
  BNp cbn2{F(), F(), F(), F()};
  const float *l1W = F(), *l1b = F();
  const float *l2W = F(), *l2b = F();
  const float *l3W = F(), *l3b = F();
  CBRp emb1{{F(), F(), F(), F()}, F()};
  CBRp emb2{{F(), F(), F(), F()}, F()};
  const float *alpha[4], *beta[4];
  Resp pos[4][2], pre[4][2];
  CBRp transfer[4];
  for (int st = 0; st < 4; ++st) {
    alpha[st] = F(); beta[st] = F();
    for (int r = 0; r < 2; ++r)
      pos[st][r] = Resp{ CBRp{{F(),F(),F(),F()}, F()}, CBRp{{F(),F(),F(),F()}, F()} };
    for (int r = 0; r < 2; ++r)
      pre[st][r] = Resp{ CBRp{{F(),F(),F(),F()}, F()}, CBRp{{F(),F(),F(),F()}, F()} };
    transfer[st] = CBRp{{F(),F(),F(),F()}, F()};
  }
  const float* pts = F();

  // ---- workspace ----------------------------------------------------------
  char* ws = (char*)d_ws; size_t off = 0;
  auto alloc = [&](size_t bytes) -> void* {
    void* p = ws + off; off = (off + bytes + 255) & ~(size_t)255; return p;
  };
  float* xyzN[2] = { (float*)alloc((size_t)BATCH*1024*3*4),
                     (float*)alloc((size_t)BATCH*1024*3*4) };
  int*   fpsIdx  = (int*)alloc((size_t)BATCH*1024*4);
  int*   knnIdx  = (int*)alloc((size_t)BATCH*1024*KNN_K*4);
  float* gmean   = (float*)alloc((size_t)573440*4);
  float* stats   = (float*)alloc((size_t)2*BATCH*4);
  float* graw    = (float*)alloc((size_t)18350080*4);
  h16*   Xg      = (h16*)alloc((size_t)50331648*2);   // max Ng*dchPad (st1: 524288*96)
  h16*   iA      = (h16*)alloc((size_t)33554432*2);
  h16*   iB      = (h16*)alloc((size_t)33554432*2);
  h16*   dat[2]  = { (h16*)alloc((size_t)1048576*2), (h16*)alloc((size_t)1048576*2) };
  h16*   dtmp    = (h16*)alloc((size_t)1048576*2);
  h16*   warena  = (h16*)alloc((size_t)8*1024*1024*2);

  // ---- weight pre-conversion (f32 -> padded f16) --------------------------
  size_t woff = 0;
  auto cvt = [&](const float* W, int M, int K) -> h16* {
    int Mp = (M + 63) & ~63, Kp = (K + 31) & ~31;
    h16* p = warena + woff; woff += (size_t)Mp*Kp;
    long long tot = (long long)Mp*Kp;
    k_cvt_pad<<<(unsigned)((tot + 255)/256), 256, 0, stream>>>(W, p, M, K, Kp, tot);
    return p;
  };

  auto gemm = [&](const h16* Wh, const h16* X, h16* Y, const float* bias,
                  const BNp* bn, const h16* resp, int Cout, int Kst, int N, int relu) {
    dim3 g(N/32, (Cout + 63)/64);
    k_gemm<<<g, 256, 0, stream>>>(Wh, X, Y, bias,
        bn ? bn->g : nullptr, bn ? bn->b : nullptr,
        bn ? bn->m : nullptr, bn ? bn->v : nullptr,
        resp, Cout, Kst, N, relu);
  };

  // ---- embedding ----------------------------------------------------------
  int P0 = BATCH * NPTS;
  k_embed<<<(P0+255)/256, 256, 0, stream>>>(pts, normals,
      emb1.W, emb1.bn.g, emb1.bn.b, emb1.bn.m, emb1.bn.v,
      emb2.W, emb2.bn.g, emb2.bn.b, emb2.bn.m, emb2.bn.v,
      dat[0], P0);

  // ---- stages -------------------------------------------------------------
  int Ncur = NPTS, C = 32;
  const float* xyz = pts;
  h16* data = dat[0];
  for (int st = 0; st < 4; ++st) {
    int S = Ncur / 2, G = BATCH * S, Cout = 2*C, dch = 2*C + 3;
    int dchPad = (dch + 31) & ~31;
    long long Mb = (long long)S * (C+3) * KNN_K;

    k_fps<<<BATCH, 256, 0, stream>>>(xyz, Ncur, S, fpsIdx);
    float* nxyz = xyzN[st & 1];
    k_gather_xyz<<<(G+255)/256, 256, 0, stream>>>(xyz, fpsIdx, nxyz, S, Ncur);
    k_knn<<<(G+255)/256, 256, 0, stream>>>(xyz, nxyz, knnIdx, S, Ncur);

    long long tg = (long long)G * (C+3);
    k_group_gather<<<(unsigned)((tg+255)/256), 256, 0, stream>>>(
        data, xyz, knnIdx, graw, gmean, S, C, Ncur);
    k_fill<<<1, 64, 0, stream>>>(stats, 2*BATCH);
    k_stats<<<(unsigned)((long long)BATCH*Mb/256), 256, 0, stream>>>(graw, gmean, stats, Mb);
    long long tn = (long long)G * KNN_K * dchPad;
    k_norm_concat<<<(unsigned)((tn+255)/256), 256, 0, stream>>>(
        graw, gmean, stats, alpha[st], beta[st], data, fpsIdx, Xg, S, C, Ncur, Mb, dchPad);

    int Ng = G * KNN_K;
    h16* tW = cvt(transfer[st].W, Cout, dch);
    gemm(tW, Xg, iA, nullptr, &transfer[st].bn, nullptr, Cout, dchPad, Ng, 1);
    for (int r = 0; r < 2; ++r) {
      h16* w1 = cvt(pre[st][r].n1.W, Cout, Cout);
      h16* w2 = cvt(pre[st][r].n2.W, Cout, Cout);
      gemm(w1, iA, iB, nullptr, &pre[st][r].n1.bn, nullptr, Cout, Cout, Ng, 1);
      gemm(w2, iB, iA, nullptr, &pre[st][r].n2.bn, iA,      Cout, Cout, Ng, 1);
    }
    h16* dnext = dat[(st+1) & 1];
    k_maxpool<<<(unsigned)(((long long)Cout*G + 255)/256), 256, 0, stream>>>(iA, dnext, Cout, G);
    for (int r = 0; r < 2; ++r) {
      h16* w1 = cvt(pos[st][r].n1.W, Cout, Cout);
      h16* w2 = cvt(pos[st][r].n2.W, Cout, Cout);
      gemm(w1, dnext, dtmp, nullptr, &pos[st][r].n1.bn, nullptr, Cout, Cout, G, 1);
      gemm(w2, dtmp, dnext, nullptr, &pos[st][r].n2.bn, dnext,   Cout, Cout, G, 1);
    }

    xyz = nxyz; data = dnext; Ncur = S; C = Cout;
  }

  // ---- classifier head ----------------------------------------------------
  int Pf = BATCH * Ncur;   // 2048 points
  h16* c1 = cvt(l1W, 256, 512);
  h16* c2 = cvt(l2W, 128, 256);
  h16* c3 = cvt(l3W, 3, 128);
  gemm(c1, data, iA, l1b, &cbn1, nullptr, 256, 512, Pf, 1);
  gemm(c2, iA,   iB, l2b, &cbn2, nullptr, 128, 256, Pf, 1);
  gemm(c3, iB, dtmp, l3b, nullptr, nullptr, 3, 128, Pf, 0);
  k_mean<<<1, 64, 0, stream>>>(dtmp, (float*)d_out, Ncur);
}
